// AdditiveAttention_1219770712716
// MI455X (gfx1250) — compile-verified, matrix operand order checked by reading, not run on hardware
//
#include <hip/hip_runtime.h>
#include <hip/hip_bf16.h>
#include <math.h>

// Problem sizes (fixed by the reference)
#define BATCH 64
#define TSEQ  2048
#define DDIM  128   // = K of the projection GEMM
#define UDIM  128   // = N of the projection GEMM

typedef __attribute__((ext_vector_type(16))) __bf16 v16bf;
typedef __attribute__((ext_vector_type(8)))  float  v8f;
typedef __attribute__((ext_vector_type(4)))  float  v4f;

// ---------------------------------------------------------------------------
// Workspace layout (bytes):
//   [0, 32768)          : Wc bf16 B-fragments  [8 u-tiles][4 k-tiles][32 lanes][16 bf16]
//   [32768, 33280)      : bc = W1_b + W2_b (128 f32)
//   [33280, 557568)     : scores [B*T] f32
//   [557568, 558080)    : stats: max[64], denom[64]
// ---------------------------------------------------------------------------
#define WS_FRAG_OFF   0
#define WS_BC_OFF     32768
#define WS_SCORE_OFF  33280
#define WS_STATS_OFF  (33280 + BATCH * TSEQ * 4)

// Branch-free tanh: native V_TANH_F32 on gfx1250 if the builtin exists.
__device__ __forceinline__ float fast_tanh(float x) {
#if __has_builtin(__builtin_amdgcn_tanhf)
    return __builtin_amdgcn_tanhf(x);
#else
    // tanh(x) = sign(x) * (1 - 2e/(1+e)), e = exp(-2|x|); branch-free, v_exp_f32
    float e = __expf(-2.0f * __builtin_fabsf(x));
    float t = 1.0f - 2.0f * e / (1.0f + e);
    return __builtin_copysignf(t, x);
#endif
}

// ===========================================================================
// Kernel 0: fold W1+W2 -> bf16, pre-swizzled into WMMA B-fragment order.
// B (32x16, KxN) 16-bit layout per lane l: column N = l%16, half = l/16,
// value i -> K = 32*kt + (i<8 ? 8*half + i : 16 + 8*half + (i-8)).
// ===========================================================================
__global__ void addattn_prep_kernel(const float* __restrict__ W1,
                                    const float* __restrict__ W2,
                                    const float* __restrict__ b1,
                                    const float* __restrict__ b2,
                                    __bf16* __restrict__ wfrag,
                                    float* __restrict__ bc) {
    int g = blockIdx.x * blockDim.x + threadIdx.x;   // 0..1023
    int j    = g >> 7;         // u-tile 0..7
    int kt   = (g >> 5) & 3;   // k-tile 0..3
    int lane = g & 31;
    int n    = 16 * j + (lane & 15);
    int half = lane >> 4;
    __bf16* dst = wfrag + (size_t)g * 16;
#pragma unroll
    for (int i = 0; i < 16; ++i) {
        int k = 32 * kt + ((i < 8) ? (8 * half + i) : (16 + 8 * half + (i - 8)));
        float w = W1[k * UDIM + n] + W2[k * UDIM + n];
        dst[i] = (__bf16)w;
    }
    if (g < UDIM) bc[g] = b1[g] + b2[g];
}

// ===========================================================================
// Kernel 1: scores[b,t] = V . tanh(x[b,t,:] @ Wc + bc)   (V_b dropped:
// softmax is shift-invariant). 256 threads = 8 waves; each wave owns 32 rows
// of t (TWO 16-row A tiles sharing each B fragment -> half the LDS traffic
// per WMMA). 64 WMMAs per wave.
// ===========================================================================
__device__ __forceinline__ v16bf a_frag_from_f32(const float* __restrict__ xrow,
                                                 int kt, int half) {
    // 16-bit A layout (16x32): lane holds row M=l%16; Ks = {8h..8h+7, 16+8h..16+8h+7}
    const float* p0 = xrow + 32 * kt + 8 * half;
    const float* p1 = p0 + 16;
    v4f x0 = *(const v4f*)(p0);
    v4f x1 = *(const v4f*)(p0 + 4);
    v4f y0 = *(const v4f*)(p1);
    v4f y1 = *(const v4f*)(p1 + 4);
    v16bf a;
    a[0]  = (__bf16)x0[0]; a[1]  = (__bf16)x0[1]; a[2]  = (__bf16)x0[2]; a[3]  = (__bf16)x0[3];
    a[4]  = (__bf16)x1[0]; a[5]  = (__bf16)x1[1]; a[6]  = (__bf16)x1[2]; a[7]  = (__bf16)x1[3];
    a[8]  = (__bf16)y0[0]; a[9]  = (__bf16)y0[1]; a[10] = (__bf16)y0[2]; a[11] = (__bf16)y0[3];
    a[12] = (__bf16)y1[0]; a[13] = (__bf16)y1[1]; a[14] = (__bf16)y1[2]; a[15] = (__bf16)y1[3];
    return a;
}

__global__ void __launch_bounds__(256)
addattn_score_kernel(const float* __restrict__ enc,
                     const __bf16* __restrict__ wfrag,
                     const float* __restrict__ bc,
                     const float* __restrict__ Vw,
                     float* __restrict__ scores) {
    __shared__ v16bf sB[8][4][32];   // 32 KB of Wc fragments
    __shared__ float sbc[UDIM];
    __shared__ float sV[UDIM];

    const int tid = threadIdx.x;
    // Stage Wc fragments into LDS (32768 B = 2048 x v4f, 8 per thread).
    {
        const v4f* src = (const v4f*)wfrag;
        v4f* dst = (v4f*)sB;
#pragma unroll
        for (int i = 0; i < 8; ++i) dst[tid + 256 * i] = src[tid + 256 * i];
        if (tid < UDIM) { sbc[tid] = bc[tid]; sV[tid] = Vw[tid]; }
    }
    __syncthreads();

    const int b     = blockIdx.x >> 3;               // 8 blocks per batch row
    const int tseg  = blockIdx.x & 7;
    const int wave  = tid >> 5;
    const int lane  = tid & 31;
    const int half  = lane >> 4;
    const int tbase = tseg * 256 + wave * 32;        // 32 t-rows per wave

    const float* xrow0 = enc + ((size_t)b * TSEQ + tbase + (lane & 15)) * DDIM;
    const float* xrow1 = xrow0 + (size_t)16 * DDIM;  // second 16-row tile

    // A fragments for both tiles, full K=128 (4 k-tiles each), f32 -> bf16.
    v16bf a0[4], a1[4];
#pragma unroll
    for (int kt = 0; kt < 4; ++kt) {
        a0[kt] = a_frag_from_f32(xrow0, kt, half);
        a1[kt] = a_frag_from_f32(xrow1, kt, half);
    }

    float sp0[8], sp1[8];
#pragma unroll
    for (int r = 0; r < 8; ++r) { sp0[r] = 0.0f; sp1[r] = 0.0f; }

#pragma unroll
    for (int j = 0; j < 8; ++j) {                    // 8 u-tiles of 16 columns
        v8f c0 = {};
        v8f c1 = {};
#pragma unroll
        for (int kt = 0; kt < 4; ++kt) {
            v16bf bf = sB[j][kt][lane];              // one LDS read, two WMMAs
            c0 = __builtin_amdgcn_wmma_f32_16x16x32_bf16(false, a0[kt], false, bf,
                                                         (short)0, c0, false, false);
            c1 = __builtin_amdgcn_wmma_f32_16x16x32_bf16(false, a1[kt], false, bf,
                                                         (short)0, c1, false, false);
        }
        const int u     = 16 * j + (lane & 15);      // C layout: lanes <-> columns
        const float bcv = sbc[u];
        const float vv  = sV[u];
#pragma unroll
        for (int r = 0; r < 8; ++r) {                // row M = r + 8*half
            sp0[r] += fast_tanh(c0[r] + bcv) * vv;
            sp1[r] += fast_tanh(c1[r] + bcv) * vv;
        }
    }

    // Reduce the 16 columns of each row across each 16-lane half.
    float* srow = scores + (size_t)b * TSEQ + tbase;
#pragma unroll
    for (int r = 0; r < 8; ++r) {
        float s = sp0[r];
        s += __shfl_xor(s, 1, 16);
        s += __shfl_xor(s, 2, 16);
        s += __shfl_xor(s, 4, 16);
        s += __shfl_xor(s, 8, 16);
        float q = sp1[r];
        q += __shfl_xor(q, 1, 16);
        q += __shfl_xor(q, 2, 16);
        q += __shfl_xor(q, 4, 16);
        q += __shfl_xor(q, 8, 16);
        if ((lane & 15) == 0) {
            srow[r + 8 * half] = s;
            srow[16 + r + 8 * half] = q;
        }
    }
}

// ===========================================================================
// Kernel 2: per-b softmax stats (max, sum of exp); also zero the output
// (harness poisons d_out, and kernel 3 accumulates with atomics).
// ===========================================================================
__global__ void addattn_stats_kernel(const float* __restrict__ scores,
                                     float* __restrict__ stats,
                                     float* __restrict__ out) {
    const int b = blockIdx.x;
    const int tid = threadIdx.x;
    __shared__ float red[256];

    float m = -INFINITY;
    for (int t = tid; t < TSEQ; t += 256)
        m = fmaxf(m, scores[(size_t)b * TSEQ + t]);
    red[tid] = m;
    __syncthreads();
    for (int s = 128; s > 0; s >>= 1) {
        if (tid < s) red[tid] = fmaxf(red[tid], red[tid + s]);
        __syncthreads();
    }
    const float bmax = red[0];
    __syncthreads();

    float e = 0.0f;
    for (int t = tid; t < TSEQ; t += 256)
        e += __expf(scores[(size_t)b * TSEQ + t] - bmax);
    red[tid] = e;
    __syncthreads();
    for (int s = 128; s > 0; s >>= 1) {
        if (tid < s) red[tid] += red[tid + s];
        __syncthreads();
    }
    if (tid == 0) { stats[b] = bmax; stats[BATCH + b] = red[0]; }
    if (tid < DDIM) out[b * DDIM + tid] = 0.0f;
}

// ===========================================================================
// Kernel 3: context[b,d] = sum_t attn[b,t] * x[b,t,d].
// 16 blocks per b, 128 t-rows each; weights in LDS, coalesced x reads,
// f32 global atomics into the tiny (32 KB) output.
// ===========================================================================
__global__ void __launch_bounds__(256)
addattn_context_kernel(const float* __restrict__ enc,
                       const float* __restrict__ scores,
                       const float* __restrict__ stats,
                       float* __restrict__ out) {
    const int b   = blockIdx.x >> 4;
    const int seg = blockIdx.x & 15;
    const int tid = threadIdx.x;
    const int g   = tid >> 7;          // 2 row-groups
    const int d   = tid & 127;

    __shared__ float w[128];
    const float bmax = stats[b];
    const float inv  = 1.0f / stats[BATCH + b];
    const int   t0   = seg * 128;
    if (tid < 128)
        w[tid] = __expf(scores[(size_t)b * TSEQ + t0 + tid] - bmax) * inv;
    __syncthreads();

    const float* xb = enc + ((size_t)b * TSEQ + t0) * DDIM;
    float acc = 0.0f;
    for (int tt = g; tt < 128; tt += 2)
        acc += w[tt] * xb[(size_t)tt * DDIM + d];

    atomicAdd(&out[b * DDIM + d], acc);
}

// ===========================================================================
extern "C" void kernel_launch(void* const* d_in, const int* in_sizes, int n_in,
                              void* d_out, int out_size, void* d_ws, size_t ws_size,
                              hipStream_t stream) {
    (void)in_sizes; (void)n_in; (void)out_size; (void)ws_size;

    const float* enc  = (const float*)d_in[0];   // [B,T,D]
    const float* W1w  = (const float*)d_in[1];   // [D,U]
    const float* W1b  = (const float*)d_in[2];   // [U]
    const float* W2w  = (const float*)d_in[3];   // [D,U]
    const float* W2b  = (const float*)d_in[4];   // [U]
    const float* Vw   = (const float*)d_in[5];   // [U,1]
    // d_in[6] = V_b: dropped (softmax shift invariance)

    char*   ws     = (char*)d_ws;
    __bf16* wfrag  = (__bf16*)(ws + WS_FRAG_OFF);
    float*  bc     = (float*)(ws + WS_BC_OFF);
    float*  scores = (float*)(ws + WS_SCORE_OFF);
    float*  stats  = (float*)(ws + WS_STATS_OFF);
    float*  out    = (float*)d_out;

    addattn_prep_kernel<<<4, 256, 0, stream>>>(W1w, W2w, W1b, W2b, wfrag, bc);

    // B*T / (8 waves * 32 rows) = 64*2048/256 = 512 blocks
    addattn_score_kernel<<<BATCH * (TSEQ / 256), 256, 0, stream>>>(
        enc, wfrag, bc, Vw, scores);

    addattn_stats_kernel<<<BATCH, 256, 0, stream>>>(scores, stats, out);

    addattn_context_kernel<<<BATCH * (TSEQ / 128), 256, 0, stream>>>(
        enc, scores, stats, out);
}
